// SelfNonParametric_Vis_70531952935517
// MI455X (gfx1250) — matrix operand order repaired and checked by reading, not compiled, not run on hardware
//
#include <hip/hip_runtime.h>
#include <hip/hip_bf16.h>
#include <math.h>

// ---------------------------------------------------------------------------
// N=16384 rows, D=128 features, C=100 classes.
// feats = concat(normalize(lb 4096), normalize(anchor 6144), normalize(pos 6144))
// sim = feats @ feats^T / 0.1 ; attn = softmax(sim) ; out = attn @ logits.
//
// Fixed-max softmax: rows are L2-normalized so sim/tau <= 10 everywhere and
// == 10 exactly on the diagonal => the row max is the constant 10. Using
// p = exp(10*s - 10) is numerically identical to max-subtracted softmax and
// eliminates all online-softmax bookkeeping. The denominator is accumulated
// by the PV GEMM itself via a ones-column appended to the padded logits.
// ---------------------------------------------------------------------------

#define N_LB   4096
#define N_ULB  6144
#define NROWS  16384
#define DIM    128
#define NC     100
#define CPAD   112          // 7 * 16 WMMA tiles; col 100 = ones, 101..111 = 0
#define WAVES  4            // waves per block in the flash kernel

typedef _Float16 half4 __attribute__((ext_vector_type(4)));
typedef _Float16 half8 __attribute__((ext_vector_type(8)));
typedef _Float16 v16h  __attribute__((ext_vector_type(16)));
typedef float    v8f   __attribute__((ext_vector_type(8)));

union V16U { v16h v; half8 h[2]; };

// ---------------------------------------------------------------------------
// Kernel 1: row L2-normalize the three feature blocks -> f16 feats[16384][128]
// One wave32 per row; each lane owns 4 contiguous floats.
// ---------------------------------------------------------------------------
__global__ __launch_bounds__(256)
void k_norm_feats(const float* __restrict__ lb,
                  const float* __restrict__ anc,
                  const float* __restrict__ pos,
                  _Float16* __restrict__ feats) {
  int wave = (int)((blockIdx.x * blockDim.x + threadIdx.x) >> 5);
  int lane = threadIdx.x & 31;
  if (wave >= NROWS) return;
  const float* src;
  if (wave < N_LB)               src = lb  + (size_t)wave * DIM;
  else if (wave < N_LB + N_ULB)  src = anc + (size_t)(wave - N_LB) * DIM;
  else                           src = pos + (size_t)(wave - N_LB - N_ULB) * DIM;

  float4 x = ((const float4*)src)[lane];
  float s = x.x * x.x + x.y * x.y + x.z * x.z + x.w * x.w;
  #pragma unroll
  for (int off = 16; off > 0; off >>= 1) s += __shfl_xor(s, off, 32);
  float inv = 1.0f / fmaxf(sqrtf(s), 1e-12f);

  half4 h;
  h[0] = (_Float16)(x.x * inv);
  h[1] = (_Float16)(x.y * inv);
  h[2] = (_Float16)(x.z * inv);
  h[3] = (_Float16)(x.w * inv);
  ((half4*)(feats + (size_t)wave * DIM))[lane] = h;
}

// ---------------------------------------------------------------------------
// Kernel 2: logits -> f16 transposed [CPAD][NROWS]; col 100 = 1.0 (row-sum
// accumulator column), cols 101..111 = 0. Also per-row argmax of
// logits_before. One wave32 per row.
// ---------------------------------------------------------------------------
__global__ __launch_bounds__(256)
void k_prep_logits(const float* __restrict__ lb,
                   const float* __restrict__ u1,
                   const float* __restrict__ u2,
                   _Float16* __restrict__ logitsT,
                   int* __restrict__ pred_before) {
  int r    = (int)((blockIdx.x * blockDim.x + threadIdx.x) >> 5);
  int lane = threadIdx.x & 31;
  if (r >= NROWS) return;
  const float* src;
  if (r < N_LB)              src = lb + (size_t)r * NC;
  else if (r < N_LB + N_ULB) src = u1 + (size_t)(r - N_LB) * NC;
  else                       src = u2 + (size_t)(r - N_LB - N_ULB) * NC;

  float best = -__builtin_inff();
  int   bidx = 0x7fffffff;
  #pragma unroll
  for (int k = 0; k < 4; ++k) {
    int c = lane + 32 * k;
    if (c < NC) {
      float v = src[c];
      logitsT[(size_t)c * NROWS + r] = (_Float16)v;
      if (v > best || (v == best && c < bidx)) { best = v; bidx = c; }
    } else if (c < CPAD) {
      // ones column feeds the softmax denominator through the PV GEMM
      logitsT[(size_t)c * NROWS + r] = (c == NC) ? (_Float16)1.0f : (_Float16)0.0f;
    }
  }
  #pragma unroll
  for (int off = 16; off > 0; off >>= 1) {
    float ov = __shfl_xor(best, off, 32);
    int   oi = __shfl_xor(bidx, off, 32);
    if (ov > best || (ov == best && oi < bidx)) { best = ov; bidx = oi; }
  }
  if (lane == 0) pred_before[r] = bidx;
}

// ---------------------------------------------------------------------------
// Kernel 3: fused attention, fixed-max softmax. One wave32 owns 16 rows.
// Per 32-column tile: 8 WMMA (QK^T, K=128) -> exp(10*s-10) -> LDS C->A layout
// transpose (per-wave scratch, s_wait_dscnt only, no block barrier) -> 7 WMMA
// (P @ [V | ones]). Denominator read out of accumulator column 100 at the end.
// ---------------------------------------------------------------------------
__global__ __launch_bounds__(32 * WAVES)
void k_attn(const _Float16* __restrict__ feats,
            const _Float16* __restrict__ logitsT,
            float* __restrict__ out_logits,   // [NROWS][NC] row-major
            int*   __restrict__ pred_after) {
  __shared__ _Float16 lds[WAVES * 16 * 32];

  int wv   = threadIdx.x >> 5;
  int lane = threadIdx.x & 31;
  int grp  = lane >> 4;       // 0: lanes 0-15, 1: lanes 16-31
  int col  = lane & 15;
  int row_base = ((int)blockIdx.x * WAVES + wv) * 16;
  _Float16* Plds = lds + wv * (16 * 32);

  // Q A-fragments (contiguous k order matching B: lanes0-15 K{q*32+0..15},
  // lanes16-31 K{+16..31}); two 16B loads per K=32 slab.
  V16U A[4];
  {
    const _Float16* qrow = feats + (size_t)(row_base + col) * DIM;
    #pragma unroll
    for (int q = 0; q < 4; ++q) {
      A[q].h[0] = *(const half8*)(qrow + 32 * q + 16 * grp);
      A[q].h[1] = *(const half8*)(qrow + 32 * q + 16 * grp + 8);
    }
  }

  v8f o[7];
  #pragma unroll
  for (int t = 0; t < 7; ++t)
    #pragma unroll
    for (int j = 0; j < 8; ++j) o[t][j] = 0.0f;

  for (int cb = 0; cb < NROWS; cb += 32) {
    // B-fragments for S: column n of K^T = feats row (cb+n) -> contiguous 32B.
    const _Float16* krow0 = feats + (size_t)(cb + col) * DIM + 16 * grp;
    const _Float16* krow1 = krow0 + (size_t)16 * DIM;
    v8f s0 = {}; v8f s1 = {};
    #pragma unroll
    for (int q = 0; q < 4; ++q) {
      v16h b0 = *(const v16h*)(krow0 + 32 * q);
      s0 = __builtin_amdgcn_wmma_f32_16x16x32_f16(false, A[q].v, false, b0,
                                                  (short)0, s0, false, false);
    }
    #pragma unroll
    for (int q = 0; q < 4; ++q) {
      v16h b1 = *(const v16h*)(krow1 + 32 * q);
      s1 = __builtin_amdgcn_wmma_f32_16x16x32_f16(false, A[q].v, false, b1,
                                                  (short)0, s1, false, false);
    }

    // p = exp(sim/tau - 10): fixed max is EXACT (diagonal == 10, C-S bound).
    // C-layout: VGPR j of lane-group g holds (row j+8g, col). Write straight
    // into the per-wave LDS transpose tile (16x32 f16, row pitch 32).
    #pragma unroll
    for (int j = 0; j < 8; ++j) {
      float p0 = __expf(__builtin_fmaf(s0[j], 10.0f, -10.0f));
      float p1 = __expf(__builtin_fmaf(s1[j], 10.0f, -10.0f));
      int r = j + 8 * grp;
      Plds[r * 32 + col]      = (_Float16)p0;
      Plds[r * 32 + col + 16] = (_Float16)p1;
    }
    // Per-wave scratch + in-order LDS pipe: only need DS-count drain, not a
    // block barrier, before the cross-lane A-layout reload.
    asm volatile("s_wait_dscnt 0" ::: "memory");
    V16U Pf;
    Pf.h[0] = *(const half8*)(Plds + col * 32 + 16 * grp);
    Pf.h[1] = *(const half8*)(Plds + col * 32 + 16 * grp + 8);

    // O += P(16x32) @ [V|1](32x112): B column n = logitsT row (16t+n).
    #pragma unroll
    for (int t = 0; t < 7; ++t) {
      v16h Bv = *(const v16h*)(logitsT + (size_t)(16 * t + col) * NROWS + cb + 16 * grp);
      o[t] = __builtin_amdgcn_wmma_f32_16x16x32_f16(false, Pf.v, false, Bv,
                                                    (short)0, o[t], false, false);
    }
  }

  // Denominator lives in accumulator column 100 = tile 6, col 4, lane 16g+4.
  #pragma unroll
  for (int j = 0; j < 8; ++j) {
    float l   = __shfl(o[6][j], 4 + 16 * grp, 32);
    float inv = 1.0f / l;
    #pragma unroll
    for (int t = 0; t < 7; ++t) o[t][j] *= inv;
  }

  // Write logits_after + argmax-after (first-max tie-break, like jnp.argmax).
  #pragma unroll
  for (int j = 0; j < 8; ++j) {
    int r = row_base + j + 8 * grp;
    float best = -__builtin_inff();
    int   bidx = 0x7fffffff;
    #pragma unroll
    for (int t = 0; t < 7; ++t) {
      int c = 16 * t + col;
      if (c < NC) {
        float v = o[t][j];
        out_logits[(size_t)r * NC + c] = v;
        if (v > best || (v == best && c < bidx)) { best = v; bidx = c; }
      }
    }
    #pragma unroll
    for (int off = 1; off < 16; off <<= 1) {
      float ov = __shfl_xor(best, off, 32);
      int   oi = __shfl_xor(bidx, off, 32);
      if (ov > best || (ov == best && oi < bidx)) { best = ov; bidx = oi; }
    }
    if (col == 0) pred_after[r] = bidx;
  }
}

// ---------------------------------------------------------------------------
// Kernel 4: scalars. softmax over a (N,1) axis is identically 1.0 -> diff == 0
// for every row, so argmax(where(changed,0,-inf)) == argmin(where(changed,0,inf))
// == first row where pred_before != pred_after (0 if none changed).
// ---------------------------------------------------------------------------
__global__ __launch_bounds__(256)
void k_finalize(const int* __restrict__ pred_before,
                const int* __restrict__ pred_after,
                float* __restrict__ out_scalars) {
  __shared__ int sh[256];
  int tid = threadIdx.x;
  int first = 0x7fffffff;
  for (int r = tid; r < NROWS; r += 256)
    if (pred_before[r] != pred_after[r] && r < first) first = r;
  sh[tid] = first;
  __syncthreads();
  for (int s = 128; s > 0; s >>= 1) {
    if (tid < s) sh[tid] = min(sh[tid], sh[tid + s]);
    __syncthreads();
  }
  if (tid == 0) {
    int idx = sh[0];
    if (idx == 0x7fffffff) idx = 0;
    out_scalars[0] = (float)idx;                 // max_diff_idx
    out_scalars[1] = (float)idx;                 // min_diff_idx
    out_scalars[2] = (float)pred_before[idx];    // pred_max_diff_bf
    out_scalars[3] = (float)pred_after[idx];     // pred_max_diff_af
    out_scalars[4] = (float)pred_before[idx];    // pred_min_diff_bf
    out_scalars[5] = (float)pred_after[idx];     // pred_min_diff_af
  }
}

// ---------------------------------------------------------------------------
extern "C" void kernel_launch(void* const* d_in, const int* in_sizes, int n_in,
                              void* d_out, int out_size, void* d_ws, size_t ws_size,
                              hipStream_t stream) {
  const float* anchor_feat   = (const float*)d_in[0];  // 6144*128
  const float* positive_feat = (const float*)d_in[1];  // 6144*128
  const float* lb_feat       = (const float*)d_in[2];  // 4096*128
  const float* lb_one_hot    = (const float*)d_in[3];  // 4096*100
  const float* logits_x_lb   = (const float*)d_in[4];  // 4096*100
  const float* logits_u1     = (const float*)d_in[5];  // 6144*100
  const float* logits_u2     = (const float*)d_in[6];  // 6144*100
  (void)in_sizes; (void)n_in; (void)out_size; (void)ws_size;

  float* out = (float*)d_out;
  // output layout (flat floats, reference tuple order)
  const size_t O_ANCHOR = 0;
  const size_t O_POS    = O_ANCHOR + (size_t)N_ULB * DIM;
  const size_t O_LBF    = O_POS + (size_t)N_ULB * DIM;
  const size_t O_ONEHOT = O_LBF + (size_t)N_LB * DIM;
  const size_t O_LOGITS = O_ONEHOT + (size_t)N_LB * NC; // 16384*100 contiguous
  const size_t O_SCAL   = O_LOGITS + (size_t)NROWS * NC;

  // workspace layout
  char* ws = (char*)d_ws;
  _Float16* feats   = (_Float16*)(ws + 0);                        // 4 MiB
  _Float16* logitsT = (_Float16*)(ws + (size_t)NROWS * DIM * 2);  // 3.5 MiB
  char* p = ws + (size_t)NROWS * DIM * 2 + (size_t)CPAD * NROWS * 2;
  int* pred_before = (int*)p;  p += (size_t)NROWS * 4;
  int* pred_after  = (int*)p;

  // pass-through outputs
  hipMemcpyAsync(out + O_ANCHOR, anchor_feat,   (size_t)N_ULB * DIM * 4, hipMemcpyDeviceToDevice, stream);
  hipMemcpyAsync(out + O_POS,    positive_feat, (size_t)N_ULB * DIM * 4, hipMemcpyDeviceToDevice, stream);
  hipMemcpyAsync(out + O_LBF,    lb_feat,       (size_t)N_LB  * DIM * 4, hipMemcpyDeviceToDevice, stream);
  hipMemcpyAsync(out + O_ONEHOT, lb_one_hot,    (size_t)N_LB  * NC  * 4, hipMemcpyDeviceToDevice, stream);

  k_norm_feats<<<NROWS / 8, 256, 0, stream>>>(lb_feat, anchor_feat, positive_feat, feats);
  k_prep_logits<<<NROWS / 8, 256, 0, stream>>>(logits_x_lb, logits_u1, logits_u2,
                                               logitsT, pred_before);
  k_attn<<<NROWS / (16 * WAVES), 32 * WAVES, 0, stream>>>(
      feats, logitsT, out + O_LOGITS, pred_after);
  k_finalize<<<1, 256, 0, stream>>>(pred_before, pred_after, out + O_SCAL);
}
